// GraphClsGGNN_56221121905124
// MI455X (gfx1250) — compile-verified, matrix operand order checked by reading, not run on hardware
//
#include <hip/hip_runtime.h>
#include <hip/hip_bf16.h>
#include <math.h>

// ---------------- constants (match reference) ----------------
#define NN   50000     // nodes
#define EE   600000    // edges
#define NB   64        // graphs
#define ANN_ 64        // annotation dim
#define DD   128       // hidden dim
#define NSTEP 5
#define NET  4         // edge types
#define NCLS 10
#define FEAT (DD + ANN_)   // 192

typedef _Float16 half_t;
typedef __attribute__((ext_vector_type(16))) _Float16 v16h;
typedef __attribute__((ext_vector_type(8)))  _Float16 v8h;
typedef __attribute__((ext_vector_type(4)))  _Float16 v4h;
typedef __attribute__((ext_vector_type(8)))  float    v8f;

// ---------------- WMMA fragment loaders (cdna5 16x16x32 f16 layout) ----------
// A (row-major MxK) and B-from-rows-of-W share the same per-lane pattern:
// lane<16 : idx=lane,    holds K = k0+[0..7] and k0+16+[0..7]
// lane>=16: idx=lane-16, holds K = k0+8+[0..7] and k0+24+[0..7]
__device__ inline v16h load_frag_h16(const half_t* __restrict__ base,
                                     int row0, int ld, int k0, int lane) {
    int idx = lane & 15;
    int kb  = (lane < 16) ? 0 : 8;
    const half_t* p = base + (size_t)(row0 + idx) * ld + k0 + kb;
    v8h lo = *reinterpret_cast<const v8h*>(p);
    v8h hi = *reinterpret_cast<const v8h*>(p + 16);
    v16h r;
#pragma unroll
    for (int i = 0; i < 8; ++i) { r[i] = lo[i]; r[8 + i] = hi[i]; }
    return r;
}

__device__ inline v16h load_frag_f32(const float* __restrict__ base,
                                     int row0, int ld, int k0, int lane) {
    int idx = lane & 15;
    int kb  = (lane < 16) ? 0 : 8;
    const float* p = base + (size_t)(row0 + idx) * ld + k0 + kb;
    float4 a0 = *reinterpret_cast<const float4*>(p);
    float4 a1 = *reinterpret_cast<const float4*>(p + 4);
    float4 b0 = *reinterpret_cast<const float4*>(p + 16);
    float4 b1 = *reinterpret_cast<const float4*>(p + 20);
    v16h r;
    r[0]=(half_t)a0.x; r[1]=(half_t)a0.y; r[2]=(half_t)a0.z; r[3]=(half_t)a0.w;
    r[4]=(half_t)a1.x; r[5]=(half_t)a1.y; r[6]=(half_t)a1.z; r[7]=(half_t)a1.w;
    r[8]=(half_t)b0.x; r[9]=(half_t)b0.y; r[10]=(half_t)b0.z; r[11]=(half_t)b0.w;
    r[12]=(half_t)b1.x; r[13]=(half_t)b1.y; r[14]=(half_t)b1.z; r[15]=(half_t)b1.w;
    return r;
}

__device__ inline v8f wmma_f16(v16h a, v16h b, v8f c) {
    return __builtin_amdgcn_wmma_f32_16x16x32_f16(false, a, false, b,
                                                  (short)0, c, false, false);
}

// ordered-uint encoding for float atomicMax
__device__ inline unsigned enc_f32(float f) {
    unsigned u = __float_as_uint(f);
    return (u & 0x80000000u) ? ~u : (u | 0x80000000u);
}
__device__ inline float dec_f32(unsigned u) {
    return (u & 0x80000000u) ? __uint_as_float(u & 0x7FFFFFFFu)
                             : __uint_as_float(~u);
}

// ---------------- setup kernels ----------------
__global__ void k_f32_to_f16(const float* __restrict__ s, half_t* __restrict__ d, int n) {
    int i = blockIdx.x * blockDim.x + threadIdx.x;
    if (i < n) d[i] = (half_t)s[i];
}

__global__ void k_init_h(const float* __restrict__ ann,
                         float* __restrict__ h, half_t* __restrict__ h16) {
    size_t i = (size_t)blockIdx.x * blockDim.x + threadIdx.x;
    if (i >= (size_t)NN * DD) return;
    int d = (int)(i & (DD - 1));
    size_t n = i >> 7;
    float v = (d < ANN_) ? ann[n * ANN_ + d] : 0.0f;
    h[i] = v;
    h16[i] = (half_t)v;
}

__global__ void k_zero_f32(float* __restrict__ p, size_t n) {
    size_t i = (size_t)blockIdx.x * blockDim.x + threadIdx.x;
    if (i < n) p[i] = 0.0f;
}

// ---------------- step kernel A: all_t16[k] = h @ W_lin[k]^T + b_lin[k] -----
// one wave per (node-tile, col-tile); the A fragment is loaded ONCE per K-step
// and reused across all 4 edge-type B matrices (4 accumulators / 16 WMMAs).
__global__ void k_type_gemm(const half_t* __restrict__ h16,
                            const half_t* __restrict__ Wlin16,
                            const float*  __restrict__ b_lin,
                            half_t* __restrict__ all_t16) {
    int wave = (blockIdx.x * blockDim.x + threadIdx.x) >> 5;
    int lane = threadIdx.x & 31;
    int tile = wave >> 3;
    if (tile >= NN / 16) return;
    int ct = wave & 7;
    int row0 = tile * 16;
    int col0 = ct * 16;

    v8f acc0 = {}, acc1 = {}, acc2 = {}, acc3 = {};
#pragma unroll
    for (int kb = 0; kb < 4; ++kb) {
        int k0 = kb * 32;
        v16h af = load_frag_h16(h16, row0, DD, k0, lane);
        v16h b0 = load_frag_h16(Wlin16 + 0 * (size_t)DD * DD, col0, DD, k0, lane);
        v16h b1 = load_frag_h16(Wlin16 + 1 * (size_t)DD * DD, col0, DD, k0, lane);
        v16h b2 = load_frag_h16(Wlin16 + 2 * (size_t)DD * DD, col0, DD, k0, lane);
        v16h b3 = load_frag_h16(Wlin16 + 3 * (size_t)DD * DD, col0, DD, k0, lane);
        acc0 = wmma_f16(af, b0, acc0);
        acc1 = wmma_f16(af, b1, acc1);
        acc2 = wmma_f16(af, b2, acc2);
        acc3 = wmma_f16(af, b3, acc3);
    }
    int n = lane & 15;
    int o = col0 + n;
    int mbase = (lane >> 4) << 3;
    float bias0 = b_lin[0 * DD + o];
    float bias1 = b_lin[1 * DD + o];
    float bias2 = b_lin[2 * DD + o];
    float bias3 = b_lin[3 * DD + o];
#pragma unroll
    for (int r = 0; r < 8; ++r) {
        size_t row = (size_t)(row0 + r + mbase);
        all_t16[(0 * (size_t)NN + row) * DD + o] = (half_t)(acc0[r] + bias0);
        all_t16[(1 * (size_t)NN + row) * DD + o] = (half_t)(acc1[r] + bias1);
        all_t16[(2 * (size_t)NN + row) * DD + o] = (half_t)(acc2[r] + bias2);
        all_t16[(3 * (size_t)NN + row) * DD + o] = (half_t)(acc3[r] + bias3);
    }
}

// ---------------- step kernel B: edge gather + segment-sum ------------------
// one wave per edge; each lane handles 4 contiguous channels
__global__ void k_edge_scatter(const half_t* __restrict__ all_t16,
                               const int* __restrict__ src,
                               const int* __restrict__ dst,
                               const int* __restrict__ et,
                               float* __restrict__ a) {
    int gid = blockIdx.x * blockDim.x + threadIdx.x;
    int e = gid >> 5;
    if (e >= EE) return;
    int lane = gid & 31;
    int s = src[e], d = dst[e], k = et[e];
    const half_t* mp = all_t16 + ((size_t)k * NN + s) * DD + lane * 4;
    v4h m = *reinterpret_cast<const v4h*>(mp);
    float* ap = a + (size_t)d * DD + lane * 4;
#pragma unroll
    for (int j = 0; j < 4; ++j) atomicAdd(ap + j, (float)m[j]);
}

// ---------------- step kernel C: fused GRU ---------------------------------
// block = one 16-node tile (8 waves); wave w handles output cols [16w,16w+16)
__global__ void k_gru(const float*  __restrict__ a,
                      const half_t* __restrict__ h16,
                      const half_t* __restrict__ Wih16,
                      const half_t* __restrict__ Whh16,
                      const float*  __restrict__ b_ih,
                      const float*  __restrict__ b_hh,
                      float* __restrict__ h,
                      half_t* __restrict__ h16_out) {
    int tile = blockIdx.x;
    int wave = threadIdx.x >> 5;
    int lane = threadIdx.x & 31;
    int row0 = tile * 16;
    int col0 = wave * 16;

    v8f cir = {}, ciz = {}, cin_ = {}, chr_ = {}, chz = {}, chn = {};
#pragma unroll
    for (int kb = 0; kb < 4; ++kb) {
        int k0 = kb * 32;
        v16h afa = load_frag_f32(a,   row0, DD, k0, lane);
        v16h afh = load_frag_h16(h16, row0, DD, k0, lane);
        v16h b0 = load_frag_h16(Wih16, 0 * DD + col0, DD, k0, lane); // reset
        v16h b1 = load_frag_h16(Wih16, 1 * DD + col0, DD, k0, lane); // update
        v16h b2 = load_frag_h16(Wih16, 2 * DD + col0, DD, k0, lane); // new
        v16h b3 = load_frag_h16(Whh16, 0 * DD + col0, DD, k0, lane);
        v16h b4 = load_frag_h16(Whh16, 1 * DD + col0, DD, k0, lane);
        v16h b5 = load_frag_h16(Whh16, 2 * DD + col0, DD, k0, lane);
        cir  = wmma_f16(afa, b0, cir);
        ciz  = wmma_f16(afa, b1, ciz);
        cin_ = wmma_f16(afa, b2, cin_);
        chr_ = wmma_f16(afh, b3, chr_);
        chz  = wmma_f16(afh, b4, chz);
        chn  = wmma_f16(afh, b5, chn);
    }
    __syncthreads();   // all waves done reading h16 rows before we overwrite

    int n = lane & 15;
    int o = col0 + n;
    int mbase = (lane >> 4) << 3;
    float bir = b_ih[o],        bhr = b_hh[o];
    float biz = b_ih[DD + o],   bhz = b_hh[DD + o];
    float bin = b_ih[2*DD + o], bhn = b_hh[2*DD + o];
#pragma unroll
    for (int r = 0; r < 8; ++r) {
        size_t row = (size_t)(row0 + r + mbase);
        float ir = cir[r] + bir, hr = chr_[r] + bhr;
        float iz = ciz[r] + biz, hz = chz[r] + bhz;
        float in_ = cin_[r] + bin, hn = chn[r] + bhn;
        float rr = 1.0f / (1.0f + expf(-(ir + hr)));
        float zz = 1.0f / (1.0f + expf(-(iz + hz)));
        float nt = tanhf(in_ + rr * hn);
        float ho = h[row * DD + o];
        float hv = (1.0f - zz) * nt + zz * ho;
        h[row * DD + o] = hv;
        h16_out[row * DD + o] = (half_t)hv;
    }
}

// ---------------- pooling / head kernels -----------------------------------
__global__ void k_gate(const float* __restrict__ h, const float* __restrict__ ann,
                       const float* __restrict__ gw, const float* __restrict__ gb,
                       float* __restrict__ gate) {
    int gid = blockIdx.x * blockDim.x + threadIdx.x;
    int n = gid >> 5;
    if (n >= NN) return;
    int lane = gid & 31;
    float s = 0.0f;
    for (int j = lane; j < DD;   j += 32) s += h[(size_t)n * DD + j] * gw[j];
    for (int j = lane; j < ANN_; j += 32) s += ann[(size_t)n * ANN_ + j] * gw[DD + j];
#pragma unroll
    for (int off = 16; off > 0; off >>= 1) s += __shfl_down(s, off, 32);
    if (lane == 0) gate[n] = s + gb[0];
}

__global__ void k_pool_init(unsigned* __restrict__ gmaxu, float* __restrict__ denom,
                            float* __restrict__ readout) {
    int i = blockIdx.x * blockDim.x + threadIdx.x;
    if (i < NB) { gmaxu[i] = 0u; denom[i] = 0.0f; }
    if (i < NB * FEAT) readout[i] = 0.0f;
}

__global__ void k_segmax(const float* __restrict__ gate, const int* __restrict__ n2g,
                         unsigned* __restrict__ gmaxu) {
    int n = blockIdx.x * blockDim.x + threadIdx.x;
    if (n >= NN) return;
    atomicMax(&gmaxu[n2g[n]], enc_f32(gate[n]));
}

__global__ void k_exdenom(const float* __restrict__ gate, const unsigned* __restrict__ gmaxu,
                          const int* __restrict__ n2g, float* __restrict__ ex,
                          float* __restrict__ denom) {
    int n = blockIdx.x * blockDim.x + threadIdx.x;
    if (n >= NN) return;
    int g = n2g[n];
    float e = expf(gate[n] - dec_f32(gmaxu[g]));
    ex[n] = e;
    atomicAdd(&denom[g], e);
}

__global__ void k_readout(const float* __restrict__ h, const float* __restrict__ ann,
                          const float* __restrict__ ex, const int* __restrict__ n2g,
                          float* __restrict__ readout) {
    size_t i = (size_t)blockIdx.x * blockDim.x + threadIdx.x;
    if (i >= (size_t)NN * FEAT) return;
    int j = (int)(i % FEAT);
    size_t n = i / FEAT;
    float f = (j < DD) ? h[n * DD + j] : ann[n * ANN_ + (j - DD)];
    atomicAdd(&readout[(size_t)n2g[n] * FEAT + j], f * ex[n]);
}

__global__ void k_final(const float* __restrict__ readout, const float* __restrict__ denom,
                        const float* __restrict__ out_w, const float* __restrict__ out_b,
                        const int* __restrict__ labels, float* __restrict__ out) {
    __shared__ float ls[NB];
    int g = threadIdx.x;           // blockDim.x == NB
    float inv = 1.0f / denom[g];
    float logits[NCLS];
#pragma unroll
    for (int c = 0; c < NCLS; ++c) {
        float s = out_b[c];
        for (int j = 0; j < FEAT; ++j)
            s += readout[g * FEAT + j] * inv * out_w[c * FEAT + j];
        logits[c] = s;
    }
    int best = 0; float bm = logits[0];
#pragma unroll
    for (int c = 1; c < NCLS; ++c) if (logits[c] > bm) { bm = logits[c]; best = c; }
    float se = 0.0f;
#pragma unroll
    for (int c = 0; c < NCLS; ++c) se += expf(logits[c] - bm);
    float lse = bm + logf(se);
    ls[g] = -(logits[labels[g]] - lse);
    out[1 + g] = (float)best;      // preds
    __syncthreads();
    if (g == 0) {
        float s = 0.0f;
        for (int i = 0; i < NB; ++i) s += ls[i];
        out[0] = s / (float)NB;    // loss
    }
}

// ---------------- host launcher --------------------------------------------
static inline size_t align256(size_t x) { return (x + 255) & ~(size_t)255; }

extern "C" void kernel_launch(void* const* d_in, const int* in_sizes, int n_in,
                              void* d_out, int out_size, void* d_ws, size_t ws_size,
                              hipStream_t stream) {
    const float* annotation = (const float*)d_in[0];
    const int*   src        = (const int*)d_in[1];
    const int*   dst        = (const int*)d_in[2];
    const int*   etypes     = (const int*)d_in[3];
    const int*   node2graph = (const int*)d_in[4];
    const int*   labels     = (const int*)d_in[5];
    const float* W_lin      = (const float*)d_in[6];
    const float* b_lin      = (const float*)d_in[7];
    const float* W_ih       = (const float*)d_in[8];
    const float* W_hh       = (const float*)d_in[9];
    const float* b_ih       = (const float*)d_in[10];
    const float* b_hh       = (const float*)d_in[11];
    const float* gate_w     = (const float*)d_in[12];
    const float* gate_b     = (const float*)d_in[13];
    const float* out_w      = (const float*)d_in[14];
    const float* out_b      = (const float*)d_in[15];

    // ---- workspace carve (~116 MB) ----
    char* p = (char*)d_ws;
    float*  h       = (float*)p;  p += align256((size_t)NN * DD * 4);
    half_t* h16     = (half_t*)p; p += align256((size_t)NN * DD * 2);
    float*  a       = (float*)p;  p += align256((size_t)NN * DD * 4);
    half_t* all_t16 = (half_t*)p; p += align256((size_t)NET * NN * DD * 2);
    half_t* Wlin16  = (half_t*)p; p += align256((size_t)NET * DD * DD * 2);
    half_t* Wih16   = (half_t*)p; p += align256((size_t)3 * DD * DD * 2);
    half_t* Whh16   = (half_t*)p; p += align256((size_t)3 * DD * DD * 2);
    float*  gate    = (float*)p;  p += align256((size_t)NN * 4);
    float*  ex      = (float*)p;  p += align256((size_t)NN * 4);
    unsigned* gmaxu = (unsigned*)p; p += align256((size_t)NB * 4);
    float*  denom   = (float*)p;  p += align256((size_t)NB * 4);
    float*  readout = (float*)p;  p += align256((size_t)NB * FEAT * 4);

    // ---- weight conversion + h init ----
    k_f32_to_f16<<<(NET * DD * DD + 255) / 256, 256, 0, stream>>>(W_lin, Wlin16, NET * DD * DD);
    k_f32_to_f16<<<(3 * DD * DD + 255) / 256, 256, 0, stream>>>(W_ih, Wih16, 3 * DD * DD);
    k_f32_to_f16<<<(3 * DD * DD + 255) / 256, 256, 0, stream>>>(W_hh, Whh16, 3 * DD * DD);
    k_init_h<<<(NN * DD + 255) / 256, 256, 0, stream>>>(annotation, h, h16);

    // ---- 5 propagation steps ----
    const int ntile = NN / 16;                        // 3125
    const int gemmA_blocks = (ntile * 8) / 8;         // 8 waves/block -> 3125
    for (int s = 0; s < NSTEP; ++s) {
        k_type_gemm<<<gemmA_blocks, 256, 0, stream>>>(h16, Wlin16, b_lin, all_t16);
        k_zero_f32<<<(NN * DD + 255) / 256, 256, 0, stream>>>(a, (size_t)NN * DD);
        k_edge_scatter<<<(EE * 32) / 256, 256, 0, stream>>>(all_t16, src, dst, etypes, a);
        k_gru<<<ntile, 256, 0, stream>>>(a, h16, Wih16, Whh16, b_ih, b_hh, h, h16);
    }

    // ---- attention pooling + classification head ----
    k_gate<<<(NN * 32 + 255) / 256, 256, 0, stream>>>(h, annotation, gate_w, gate_b, gate);
    k_pool_init<<<(NB * FEAT + 255) / 256, 256, 0, stream>>>(gmaxu, denom, readout);
    k_segmax<<<(NN + 255) / 256, 256, 0, stream>>>(gate, node2graph, gmaxu);
    k_exdenom<<<(NN + 255) / 256, 256, 0, stream>>>(gate, gmaxu, node2graph, ex, denom);
    k_readout<<<(int)(((size_t)NN * FEAT + 255) / 256), 256, 0, stream>>>(h, annotation, ex, node2graph, readout);
    k_final<<<1, NB, 0, stream>>>(readout, denom, out_w, out_b, labels, (float*)d_out);
}
